// CO2RRGNN_34273839022156
// MI455X (gfx1250) — matrix operand order repaired and checked by reading, not compiled, not run on hardware
//
#include <hip/hip_runtime.h>

// ---------------------------------------------------------------------------
// CDNA5 (gfx1250) SchNet-style GNN, bf16 WMMA pipeline.
// ---------------------------------------------------------------------------

typedef __attribute__((ext_vector_type(16))) __bf16 v16bf;
typedef __attribute__((ext_vector_type(8)))  float  v8f;

static __device__ __forceinline__ v8f wmma_bf16(v16bf a, v16bf b, v8f c) {
  // D = A(16x32) * B(32x16) + C(16x16), f32 accumulate.
  return __builtin_amdgcn_wmma_f32_16x16x32_bf16(
      /*neg_a=*/false, a, /*neg_b=*/false, b,
      /*c_mod=*/(short)0, c, /*reuse_a=*/false, /*reuse_b=*/false);
}

// A-matrix (16-bit, 16x32) K pattern within a 32-wide k-tile:
// lanes 0-15 (h=0): VGPR0..3 -> K 0..7, VGPR4..7 -> K 16..23
// lanes 16-31 (h=1): K 8..15 and 24..31
static __device__ __forceinline__ int a_pat(int h, int j) {
  return (j < 8) ? (8 * h + j) : (8 * h + j + 8);
}

// Wave-local LDS ordering: CDNA5 split dependency counter wait.
static __device__ __forceinline__ void wave_lds_sync() {
  __builtin_amdgcn_wave_barrier();
  asm volatile("s_wait_dscnt 0" ::: "memory");
  __builtin_amdgcn_wave_barrier();
}

// Scatter a 16x128 C-layout result (8 n-tiles of v8f) into LDS laid out in
// A-fragment order for the next GEMM (4 k-tiles of 16x32 bf16).
static __device__ __forceinline__ void stage_c_to_a(__bf16* stage,
                                                    const v8f* z, int lane) {
  const int lm = lane & 15, h = lane >> 4;
#pragma unroll
  for (int nt = 0; nt < 8; ++nt) {
    const int n = nt * 16 + lm;        // column of C == K index of next A
    const int kt2 = n >> 5;
    const int kk = n & 31;
    const int h2 = (kk >> 3) & 1;
    const int j = (kk & 7) + ((kk & 16) ? 8 : 0);
#pragma unroll
    for (int r = 0; r < 8; ++r) {
      const int mrow = r + 8 * h;      // row of C == M of next A
      stage[kt2 * 512 + (h2 * 16 + mrow) * 16 + j] = (__bf16)z[nt][r];
    }
  }
}

// ---------------------------------------------------------------------------
// Weight pre-pack: f32 [K,N] row-major -> bf16 B-fragment order.
// frag = kt*(N/16)+nt ; element index = lane*16+j ; k = kt*32 + a_pat(lane/16, j)
// ---------------------------------------------------------------------------
__global__ void prep_bfrag_kernel(const float* __restrict__ src,
                                  unsigned short* __restrict__ dst,
                                  int K, int Ncols) {
  const int idx = blockIdx.x * blockDim.x + threadIdx.x;
  const int total = K * Ncols;
  if (idx >= total) return;
  const int frag = idx >> 9;
  const int rem = idx & 511;
  const int lane = rem >> 4, j = rem & 15;
  const int NT = Ncols >> 4;
  const int kt = frag / NT, nt = frag - kt * NT;
  const int hh = lane >> 4;
  const int n = nt * 16 + (lane & 15);
  const int k = kt * 32 + a_pat(hh, j);
  __bf16 v = (__bf16)src[k * Ncols + n];
  dst[idx] = *reinterpret_cast<unsigned short*>(&v);
}

// ---------------------------------------------------------------------------
// h = embed_w[atomic_numbers] + node_features @ proj_w + proj_b
// ---------------------------------------------------------------------------
__global__ void embed_kernel(const int* __restrict__ an,
                             const float* __restrict__ nf,
                             const float* __restrict__ ew,
                             const float* __restrict__ pw,
                             const float* __restrict__ pb,
                             float* __restrict__ hbuf, int N) {
  const int idx = blockIdx.x * blockDim.x + threadIdx.x;
  if (idx >= N * 128) return;
  const int node = idx >> 7, c = idx & 127;
  float acc = pb[c] + ew[an[node] * 128 + c];
  const float* nfr = nf + (size_t)node * 64;
#pragma unroll 8
  for (int k = 0; k < 64; ++k) acc += nfr[k] * pw[k * 128 + c];
  hbuf[idx] = acc;
}

// in-degree counts (clamped at use site)
__global__ void degree_kernel(const int* __restrict__ ei,
                              float* __restrict__ cnt, int E) {
  const int e = blockIdx.x * blockDim.x + threadIdx.x;
  if (e < E) atomicAdd(&cnt[ei[E + e]], 1.0f);
}

// ---------------------------------------------------------------------------
// Per-layer edge kernel: RBF -> 3-GEMM filter MLP (WMMA) -> W*h[src] scatter.
// One wave per 16-edge tile; 4 waves/block; 16 KB dynamic LDS staging.
// ---------------------------------------------------------------------------
__global__ __launch_bounds__(128) void filter_scatter_kernel(
    const int* __restrict__ ei, const float* __restrict__ edist,
    const float* __restrict__ hbuf, const unsigned short* __restrict__ wb_raw,
    const float* __restrict__ fb1, const float* __restrict__ fb2,
    const float* __restrict__ fb3, float* __restrict__ agg, int E) {
  extern __shared__ __align__(32) char smem_raw[];
  __bf16* smem = (__bf16*)smem_raw;
  const __bf16* wb = (const __bf16*)wb_raw;

  const int lane = threadIdx.x & 31;
  const int wave = threadIdx.x >> 5;
  const int lm = lane & 15, h = lane >> 4;
  __bf16* stage = smem + wave * 2048;  // 16x128 bf16 per wave

  const int nTiles = E >> 4;
  const int wg = blockIdx.x * 4 + wave;
  const int wstride = gridDim.x * 4;

  for (int tile = wg; tile < nTiles; tile += wstride) {
    const int ebase = tile * 16;

    // ---- RBF tile 16x64 directly into two 16x32 A-fragments -------------
    const float d = edist[ebase + lm];  // row M = lm
    float cf = 0.0f;
    if (d < 6.0f) cf = 0.5f * (__cosf(d * 0.52359878f) + 1.0f);
    v16bf a0[2];
#pragma unroll
    for (int kt = 0; kt < 2; ++kt) {
#pragma unroll
      for (int j = 0; j < 16; ++j) {
        const int k = kt * 32 + a_pat(h, j);
        const float diff = d - (float)k * (6.0f / 63.0f);
        a0[kt][j] = (__bf16)(__expf(-10.0f * diff * diff) * cf);
      }
    }

    // ---- GEMM1: [16x64] @ fw1[64x128] + fb1, relu -----------------------
    v8f z[8];
#pragma unroll
    for (int nt = 0; nt < 8; ++nt) {
      const float bv = fb1[nt * 16 + lm];
      v8f acc;
#pragma unroll
      for (int r = 0; r < 8; ++r) acc[r] = bv;
#pragma unroll
      for (int kt = 0; kt < 2; ++kt) {
        v16bf b = *(const v16bf*)(wb + (kt * 8 + nt) * 512 + lane * 16);
        acc = wmma_bf16(a0[kt], b, acc);
      }
#pragma unroll
      for (int r = 0; r < 8; ++r) acc[r] = fmaxf(acc[r], 0.0f);
      z[nt] = acc;
    }
    stage_c_to_a(stage, z, lane);
    wave_lds_sync();
    v16bf a1[4];
#pragma unroll
    for (int kt = 0; kt < 4; ++kt)
      a1[kt] = *(const v16bf*)(stage + kt * 512 + lane * 16);
    wave_lds_sync();

    // edge endpoints for this lane's 8 C rows (m = r + 8h); prefetch gathers
    int sidx[8], didx[8];
#pragma unroll
    for (int r = 0; r < 8; ++r) {
      const int e = ebase + r + 8 * h;
      sidx[r] = ei[e];
      didx[r] = ei[E + e];
      __builtin_prefetch(hbuf + (size_t)sidx[r] * 128, 0, 0);
    }

    // ---- GEMM2: @ fw2[128x128] + fb2, relu ------------------------------
#pragma unroll
    for (int nt = 0; nt < 8; ++nt) {
      const float bv = fb2[nt * 16 + lm];
      v8f acc;
#pragma unroll
      for (int r = 0; r < 8; ++r) acc[r] = bv;
#pragma unroll
      for (int kt = 0; kt < 4; ++kt) {
        v16bf b = *(const v16bf*)(wb + 8192 + (kt * 8 + nt) * 512 + lane * 16);
        acc = wmma_bf16(a1[kt], b, acc);
      }
#pragma unroll
      for (int r = 0; r < 8; ++r) acc[r] = fmaxf(acc[r], 0.0f);
      z[nt] = acc;
    }
    stage_c_to_a(stage, z, lane);
    wave_lds_sync();
    v16bf a2[4];
#pragma unroll
    for (int kt = 0; kt < 4; ++kt)
      a2[kt] = *(const v16bf*)(stage + kt * 512 + lane * 16);
    wave_lds_sync();

    // ---- GEMM3: @ fw3[128x128] + fb3 (no relu) --------------------------
#pragma unroll
    for (int nt = 0; nt < 8; ++nt) {
      const float bv = fb3[nt * 16 + lm];
      v8f acc;
#pragma unroll
      for (int r = 0; r < 8; ++r) acc[r] = bv;
#pragma unroll
      for (int kt = 0; kt < 4; ++kt) {
        v16bf b = *(const v16bf*)(wb + 24576 + (kt * 8 + nt) * 512 + lane * 16);
        acc = wmma_bf16(a2[kt], b, acc);
      }
      z[nt] = acc;
    }

    // ---- messages = W * h[src]; scatter-add to agg[dst] (f32, L2) -------
#pragma unroll
    for (int nt = 0; nt < 8; ++nt) {
      const int col = nt * 16 + lm;
#pragma unroll
      for (int r = 0; r < 8; ++r) {
        const float hv = hbuf[(size_t)sidx[r] * 128 + col];
        atomicAdd(agg + (size_t)didx[r] * 128 + col, z[nt][r] * hv);
      }
    }
  }
}

// ---------------------------------------------------------------------------
// Per-layer node kernel: [h | agg/cnt] @ uw1 -> relu -> @ uw2 -> LN -> h +=
// One wave per 16-node tile.
// ---------------------------------------------------------------------------
__global__ __launch_bounds__(128) void update_kernel(
    float* __restrict__ hbuf, const float* __restrict__ agg,
    const float* __restrict__ cnt, const unsigned short* __restrict__ wu1_raw,
    const unsigned short* __restrict__ wu2_raw, const float* __restrict__ ub1,
    const float* __restrict__ ub2, const float* __restrict__ lng,
    const float* __restrict__ lnb, int N) {
  extern __shared__ __align__(32) char smem_raw[];
  __bf16* smem = (__bf16*)smem_raw;
  const __bf16* wu1 = (const __bf16*)wu1_raw;
  const __bf16* wu2 = (const __bf16*)wu2_raw;

  const int lane = threadIdx.x & 31;
  const int wave = threadIdx.x >> 5;
  const int lm = lane & 15, h = lane >> 4;
  __bf16* stage = smem + wave * 2048;

  const int nTiles = N >> 4;
  const int wg = blockIdx.x * 4 + wave;
  const int wstride = gridDim.x * 4;

  for (int tile = wg; tile < nTiles; tile += wstride) {
    const int node_a = tile * 16 + lm;  // A row = lm
    const float inv = 1.0f / fmaxf(cnt[node_a], 1.0f);

    // A fragments of combined[16 x 256] = [h | agg/cnt]
    v16bf a[8];
#pragma unroll
    for (int kt = 0; kt < 8; ++kt) {
#pragma unroll
      for (int j = 0; j < 16; ++j) {
        const int k = kt * 32 + a_pat(h, j);
        const float v = (k < 128)
                            ? hbuf[(size_t)node_a * 128 + k]
                            : agg[(size_t)node_a * 128 + (k - 128)] * inv;
        a[kt][j] = (__bf16)v;
      }
    }

    // GEMM1: [16x256] @ uw1[256x128] + ub1, relu
    v8f z[8];
#pragma unroll
    for (int nt = 0; nt < 8; ++nt) {
      const float bv = ub1[nt * 16 + lm];
      v8f acc;
#pragma unroll
      for (int r = 0; r < 8; ++r) acc[r] = bv;
#pragma unroll
      for (int kt = 0; kt < 8; ++kt) {
        v16bf b = *(const v16bf*)(wu1 + (kt * 8 + nt) * 512 + lane * 16);
        acc = wmma_bf16(a[kt], b, acc);
      }
#pragma unroll
      for (int r = 0; r < 8; ++r) acc[r] = fmaxf(acc[r], 0.0f);
      z[nt] = acc;
    }
    stage_c_to_a(stage, z, lane);
    wave_lds_sync();
    v16bf a2[4];
#pragma unroll
    for (int kt = 0; kt < 4; ++kt)
      a2[kt] = *(const v16bf*)(stage + kt * 512 + lane * 16);
    wave_lds_sync();

    // GEMM2: @ uw2[128x128] + ub2
    v8f u[8];
#pragma unroll
    for (int nt = 0; nt < 8; ++nt) {
      const float bv = ub2[nt * 16 + lm];
      v8f acc;
#pragma unroll
      for (int r = 0; r < 8; ++r) acc[r] = bv;
#pragma unroll
      for (int kt = 0; kt < 4; ++kt) {
        v16bf b = *(const v16bf*)(wu2 + (kt * 8 + nt) * 512 + lane * 16);
        acc = wmma_bf16(a2[kt], b, acc);
      }
      u[nt] = acc;
    }

    // LayerNorm per row (128 cols spread over 16 lanes x 8 regs)
    float mu[8], rs[8];
#pragma unroll
    for (int r = 0; r < 8; ++r) {
      float s1 = 0.0f, s2 = 0.0f;
#pragma unroll
      for (int nt = 0; nt < 8; ++nt) {
        const float x = u[nt][r];
        s1 += x;
        s2 += x * x;
      }
#pragma unroll
      for (int mask = 1; mask <= 8; mask <<= 1) {
        s1 += __shfl_xor(s1, mask, 32);
        s2 += __shfl_xor(s2, mask, 32);
      }
      const float m_ = s1 * (1.0f / 128.0f);
      const float var = s2 * (1.0f / 128.0f) - m_ * m_;
      mu[r] = m_;
      rs[r] = rsqrtf(var + 1e-5f);
    }

    // residual: h += LN(u)*g + b
#pragma unroll
    for (int nt = 0; nt < 8; ++nt) {
      const int col = nt * 16 + lm;
      const float g = lng[col], bb = lnb[col];
#pragma unroll
      for (int r = 0; r < 8; ++r) {
        const int nm = tile * 16 + r + 8 * h;
        const size_t idx = (size_t)nm * 128 + col;
        hbuf[idx] = hbuf[idx] + (u[nt][r] - mu[r]) * rs[r] * g + bb;
      }
    }
  }
}

// ---------------------------------------------------------------------------
// Output heads: atom_E -> energy segment sum; atom_S -> batch mean numerators.
// One block (64 threads) per node.
// ---------------------------------------------------------------------------
__global__ __launch_bounds__(64) void heads_kernel(
    const float* __restrict__ hbuf, const int* __restrict__ batch,
    const float* __restrict__ ew1, const float* __restrict__ eb1,
    const float* __restrict__ ew2, const float* __restrict__ eb2,
    const float* __restrict__ sw1, const float* __restrict__ sb1,
    const float* __restrict__ sw2, const float* __restrict__ sb2,
    float* __restrict__ energy, float* __restrict__ bsum,
    float* __restrict__ bcnt, int N) {
  const int node = blockIdx.x;
  if (node >= N) return;
  const int j = threadIdx.x;  // 0..63
  const float* hr = hbuf + (size_t)node * 128;
  float he = eb1[j], hs = sb1[j];
#pragma unroll 4
  for (int c = 0; c < 128; ++c) {
    const float hv = hr[c];
    he += hv * ew1[c * 64 + j];
    hs += hv * sw1[c * 64 + j];
  }
  he = fmaxf(he, 0.0f);
  hs = fmaxf(hs, 0.0f);
  float pe = he * ew2[j];
  float p0 = hs * sw2[j * 4 + 0];
  float p1 = hs * sw2[j * 4 + 1];
  float p2 = hs * sw2[j * 4 + 2];
  float p3 = hs * sw2[j * 4 + 3];
#pragma unroll
  for (int mask = 16; mask >= 1; mask >>= 1) {
    pe += __shfl_xor(pe, mask, 32);
    p0 += __shfl_xor(p0, mask, 32);
    p1 += __shfl_xor(p1, mask, 32);
    p2 += __shfl_xor(p2, mask, 32);
    p3 += __shfl_xor(p3, mask, 32);
  }
  __shared__ float red[2][5];
  const int wv = threadIdx.x >> 5, ln = threadIdx.x & 31;
  if (ln == 0) {
    red[wv][0] = pe; red[wv][1] = p0; red[wv][2] = p1;
    red[wv][3] = p2; red[wv][4] = p3;
  }
  __syncthreads();
  if (threadIdx.x == 0) {
    const int b = batch[node];
    atomicAdd(&energy[b], red[0][0] + red[1][0] + eb2[0]);
    atomicAdd(&bsum[b * 4 + 0], red[0][1] + red[1][1] + sb2[0]);
    atomicAdd(&bsum[b * 4 + 1], red[0][2] + red[1][2] + sb2[1]);
    atomicAdd(&bsum[b * 4 + 2], red[0][3] + red[1][3] + sb2[2]);
    atomicAdd(&bsum[b * 4 + 3], red[0][4] + red[1][4] + sb2[3]);
    atomicAdd(&bcnt[b], 1.0f);
  }
}

__global__ void selectivity_kernel(const float* __restrict__ bsum,
                                   const float* __restrict__ bcnt,
                                   float* __restrict__ sel, int B) {
  const int idx = blockIdx.x * blockDim.x + threadIdx.x;
  if (idx >= B * 4) return;
  sel[idx] = bsum[idx] / fmaxf(bcnt[idx >> 2], 1.0f);
}

// ---------------------------------------------------------------------------
extern "C" void kernel_launch(void* const* d_in, const int* in_sizes, int n_in,
                              void* d_out, int out_size, void* d_ws,
                              size_t ws_size, hipStream_t stream) {
  (void)n_in; (void)ws_size;
  const int*   an     = (const int*)d_in[0];
  const float* nf     = (const float*)d_in[1];
  const int*   ei     = (const int*)d_in[2];
  const float* edist  = (const float*)d_in[3];
  const int*   batch  = (const int*)d_in[4];
  const float* embw   = (const float*)d_in[5];
  const float* projw  = (const float*)d_in[6];
  const float* projb  = (const float*)d_in[7];
  const float* fw1    = (const float*)d_in[8];
  const float* fb1    = (const float*)d_in[9];
  const float* fw2    = (const float*)d_in[10];
  const float* fb2    = (const float*)d_in[11];
  const float* fw3    = (const float*)d_in[12];
  const float* fb3    = (const float*)d_in[13];
  const float* uw1    = (const float*)d_in[14];
  const float* ub1    = (const float*)d_in[15];
  const float* uw2    = (const float*)d_in[16];
  const float* ub2    = (const float*)d_in[17];
  const float* lng    = (const float*)d_in[18];
  const float* lnb    = (const float*)d_in[19];
  const float* ew1    = (const float*)d_in[20];
  const float* eb1    = (const float*)d_in[21];
  const float* ew2    = (const float*)d_in[22];
  const float* eb2    = (const float*)d_in[23];
  const float* sw1    = (const float*)d_in[24];
  const float* sb1    = (const float*)d_in[25];
  const float* sw2    = (const float*)d_in[26];
  const float* sb2    = (const float*)d_in[27];

  const int N = in_sizes[0];
  const int E = in_sizes[3];
  const int B = (out_size - N * 128) / 5;

  float* out    = (float*)d_out;
  float* energy = out;            // [B]
  float* sel    = out + B;        // [B,4]
  float* hbuf   = out + 5 * B;    // [N,128] (final h lives here; updated in place)

  // workspace carve-out
  char* ws = (char*)d_ws;
  size_t off = 0;
  auto carve = [&](size_t bytes) -> void* {
    void* p = ws + off;
    off = (off + bytes + 255) & ~(size_t)255;
    return p;
  };
  float* agg  = (float*)carve((size_t)N * 128 * sizeof(float));
  float* cnt  = (float*)carve((size_t)N * sizeof(float));
  float* bsum = (float*)carve((size_t)B * 4 * sizeof(float));
  float* bcnt = (float*)carve((size_t)B * sizeof(float));
  const size_t LSTRIDE = 90112;  // bf16 elems per layer of packed weights
  unsigned short* wbf = (unsigned short*)carve(3 * LSTRIDE * sizeof(unsigned short));

  // --- pack weights into bf16 B-fragment order -----------------------------
  auto prep = [&](const float* src, unsigned short* dst, int K, int Ncols) {
    const int tot = K * Ncols;
    prep_bfrag_kernel<<<(tot + 255) / 256, 256, 0, stream>>>(src, dst, K, Ncols);
  };
  for (int l = 0; l < 3; ++l) {
    unsigned short* base = wbf + (size_t)l * LSTRIDE;
    prep(fw1 + (size_t)l * 64 * 128,  base + 0,     64, 128);
    prep(fw2 + (size_t)l * 128 * 128, base + 8192,  128, 128);
    prep(fw3 + (size_t)l * 128 * 128, base + 24576, 128, 128);
    prep(uw1 + (size_t)l * 256 * 128, base + 40960, 256, 128);
    prep(uw2 + (size_t)l * 128 * 128, base + 73728, 128, 128);
  }

  // --- zero accumulators (capture-legal) -----------------------------------
  hipMemsetAsync(out, 0, (size_t)5 * B * sizeof(float), stream);
  hipMemsetAsync(cnt, 0, (size_t)N * sizeof(float), stream);
  hipMemsetAsync(bsum, 0, (size_t)B * 4 * sizeof(float), stream);
  hipMemsetAsync(bcnt, 0, (size_t)B * sizeof(float), stream);

  // --- embedding + in-degree -----------------------------------------------
  embed_kernel<<<(N * 128 + 255) / 256, 256, 0, stream>>>(an, nf, embw, projw,
                                                          projb, hbuf, N);
  degree_kernel<<<(E + 255) / 256, 256, 0, stream>>>(ei, cnt, E);

  // --- interaction layers ---------------------------------------------------
  const int fTiles = E >> 4;
  const int fBlocks = (fTiles + 3) / 4 < 2500 ? (fTiles + 3) / 4 : 2500;
  const int uTiles = N >> 4;
  const int uBlocks = (uTiles + 3) / 4;
  const size_t lds = 4 * 2048 * sizeof(unsigned short);  // 16 KB staging
  for (int l = 0; l < 3; ++l) {
    unsigned short* base = wbf + (size_t)l * LSTRIDE;
    hipMemsetAsync(agg, 0, (size_t)N * 128 * sizeof(float), stream);
    filter_scatter_kernel<<<fBlocks, 128, lds, stream>>>(
        ei, edist, hbuf, base, fb1 + l * 128, fb2 + l * 128, fb3 + l * 128,
        agg, E);
    update_kernel<<<uBlocks, 128, lds, stream>>>(
        hbuf, agg, cnt, base + 40960, base + 73728, ub1 + l * 128,
        ub2 + l * 128, lng + l * 128, lnb + l * 128, N);
  }

  // --- output heads ---------------------------------------------------------
  heads_kernel<<<N, 64, 0, stream>>>(hbuf, batch, ew1, eb1, ew2, eb2, sw1, sb1,
                                     sw2, sb2, energy, bsum, bcnt, N);
  selectivity_kernel<<<(B * 4 + 63) / 64, 64, 0, stream>>>(bsum, bcnt, sel, B);
}
